// GIN_64106681860687
// MI455X (gfx1250) — matrix operand order repaired and checked by reading, not compiled, not run on hardware
//
#include <hip/hip_runtime.h>

#define N_NODES 100000
#define N_EDGES 1600000
#define DIM     128
#define N_GRAPHS 512
#define N_OUT   16
#define N_LAYERS 4
#define BN_EPS  1e-5f

typedef __attribute__((ext_vector_type(16))) __bf16 v16bf;
typedef __attribute__((ext_vector_type(8)))  float  v8f;

__device__ __forceinline__ unsigned short f2bf(float f) {
  unsigned int u = __float_as_uint(f);
  u += 0x7FFFu + ((u >> 16) & 1u);            // round-to-nearest-even
  return (unsigned short)(u >> 16);
}

// ---------------- elementwise helpers ----------------

__global__ void k_zero(float* __restrict__ p, int n) {
  int i = blockIdx.x * blockDim.x + threadIdx.x;
  if (i < n) p[i] = 0.0f;
}

__global__ void k_copy4(const float4* __restrict__ s, float4* __restrict__ d, int n4) {
  int i = blockIdx.x * blockDim.x + threadIdx.x;
  if (i < n4) d[i] = s[i];
}

__global__ void k_f32_to_bf16(const float* __restrict__ x, unsigned short* __restrict__ y, int n4) {
  int i = blockIdx.x * blockDim.x + threadIdx.x;
  if (i >= n4) return;
  float4 v = *(const float4*)(x + (size_t)i * 4);
  ushort4 o;
  o.x = f2bf(v.x); o.y = f2bf(v.y); o.z = f2bf(v.z); o.w = f2bf(v.w);
  *(ushort4*)(y + (size_t)i * 4) = o;
}

// ---------------- edge aggregation: agg[dst] += h[src] ----------------
// one wave per edge, lane covers 4 floats; h/agg are L2-resident (51.2MB each)

__global__ void k_edge_agg(const float* __restrict__ h, const int* __restrict__ src,
                           const int* __restrict__ dst, float* __restrict__ agg) {
  int e = blockIdx.x * 8 + (threadIdx.x >> 5);
  int lane = threadIdx.x & 31;
  if (e >= N_EDGES) return;
  int s = src[e], d = dst[e];
  float4 v = *(const float4*)(h + (size_t)s * DIM + lane * 4);
  float* b = agg + (size_t)d * DIM + lane * 4;
  atomicAdd(b + 0, v.x); atomicAdd(b + 1, v.y);
  atomicAdd(b + 2, v.z); atomicAdd(b + 3, v.w);
}

// ---------------- weight pre-swizzle into WMMA B-fragment order ----------------
// wpk[mat][ntile(8)][ktile(4)][lane(32)][dword(8)], B 32x16 bf16 fragment:
// lane<16 holds K=kb..kb+15 (2/VGPR), lane>=16 holds K=kb+16..kb+31; col = lane&15

__global__ void k_prep_w(const float* __restrict__ W1, const float* __restrict__ W2,
                         unsigned int* __restrict__ wpk) {
  int idx = blockIdx.x * blockDim.x + threadIdx.x;
  if (idx >= 8 * 8192) return;
  int j    =  idx        & 7;
  int lane = (idx >> 3)  & 31;
  int kt   = (idx >> 8)  & 3;
  int nt   = (idx >> 10) & 7;
  int m    =  idx >> 13;
  const float* src = ((m & 1) ? W2 : W1) + (size_t)(m >> 1) * DIM * DIM;
  int n  = nt * 16 + (lane & 15);
  int K0 = kt * 32 + ((lane & 16) ? 16 : 0) + 2 * j;
  unsigned int lo = f2bf(src[(size_t)K0 * DIM + n]);
  unsigned int hi = f2bf(src[(size_t)(K0 + 1) * DIM + n]);
  wpk[idx] = lo | (hi << 16);
}

// ---------------- WMMA GEMM: C[N,128] = A[N,128](bf16) @ B[128,128](bf16) ----------------
// one wave = 16 rows x 128 cols; 4 A-frags held, 8 n-tiles x 4 k-steps of wmma

__global__ __launch_bounds__(256) void k_gemm_bf16(const unsigned short* __restrict__ A,
                                                   const unsigned int* __restrict__ Bpk,
                                                   float* __restrict__ C, int nRowTiles) {
  int wave = threadIdx.x >> 5;
  int lane = threadIdx.x & 31;
  int rt = blockIdx.x * 8 + wave;
  if (rt >= nRowTiles) return;

  union AF { v16bf v; uint4 q[2]; };
  AF a[4];
  const int khalf = (lane & 16) ? 16 : 0;         // A 16x32 bf16 layout half-select
  const unsigned short* arow = A + (size_t)(rt * 16 + (lane & 15)) * DIM;
#pragma unroll
  for (int kt = 0; kt < 4; ++kt) {
    a[kt].q[0] = *(const uint4*)(arow + kt * 32 + khalf);        // K=c..c+7
    a[kt].q[1] = *(const uint4*)(arow + kt * 32 + khalf + 8);    // K=c+8..c+15
  }

#pragma unroll
  for (int nt = 0; nt < 8; ++nt) {
    v8f acc = {};
#pragma unroll
    for (int kt = 0; kt < 4; ++kt) {
      union BF { v16bf v; uint4 q[2]; } b;
      const unsigned int* bp = Bpk + (size_t)((nt * 4 + kt) * 32 + lane) * 8;
      b.q[0] = *(const uint4*)bp;
      b.q[1] = *(const uint4*)(bp + 4);
      acc = __builtin_amdgcn_wmma_f32_16x16x32_bf16(
          false, a[kt].v, false, b.v, (short)0, acc, false, false);
    }
    // C layout: VGPR r -> row rt*16 + (lane>>4)*8 + r, col nt*16 + (lane&15)
    float* crow = C + (size_t)(rt * 16 + (lane >> 4) * 8) * DIM + nt * 16 + (lane & 15);
#pragma unroll
    for (int r = 0; r < 8; ++r) crow[(size_t)r * DIM] = acc[r];
  }
}

// ---------------- BatchNorm stats / finalize / apply ----------------

__global__ void k_bn_stats(const float* __restrict__ x, float* __restrict__ stats) {
  int c = threadIdx.x;                  // 128 threads: one column each
  float s = 0.0f, sq = 0.0f;
  for (int r = blockIdx.x; r < N_NODES; r += gridDim.x) {
    float v = x[(size_t)r * DIM + c];
    s += v; sq += v * v;
  }
  atomicAdd(&stats[c], s);
  atomicAdd(&stats[DIM + c], sq);
}

__global__ void k_bn_finalize(const float* __restrict__ stats, const float* __restrict__ gamma,
                              const float* __restrict__ beta, float* __restrict__ ss) {
  int c = threadIdx.x;
  float inv_n = 1.0f / (float)N_NODES;
  float mean = stats[c] * inv_n;
  float var  = stats[DIM + c] * inv_n - mean * mean;
  float scale = gamma[c] * rsqrtf(var + BN_EPS);
  ss[c]       = scale;
  ss[DIM + c] = beta[c] - mean * scale;
}

// inner BN: relu + bf16 downconvert (feeds next GEMM only)
__global__ void k_bn_relu_bf16(const float* __restrict__ x, const float* __restrict__ ss,
                               unsigned short* __restrict__ y, int n4) {
  int i = blockIdx.x * blockDim.x + threadIdx.x;
  if (i >= n4) return;
  int c0 = (i * 4) & (DIM - 1);
  float4 v = *(const float4*)(x + (size_t)i * 4);
  ushort4 o;
  o.x = f2bf(fmaxf(0.0f, v.x * ss[c0 + 0] + ss[DIM + c0 + 0]));
  o.y = f2bf(fmaxf(0.0f, v.y * ss[c0 + 1] + ss[DIM + c0 + 1]));
  o.z = f2bf(fmaxf(0.0f, v.z * ss[c0 + 2] + ss[DIM + c0 + 2]));
  o.w = f2bf(fmaxf(0.0f, v.w * ss[c0 + 3] + ss[DIM + c0 + 3]));
  *(ushort4*)(y + (size_t)i * 4) = o;
}

// outer BN: relu -> fp32 h, fused graph-pool atomics
__global__ void k_bn_relu_pool(const float* __restrict__ x, const float* __restrict__ ss,
                               const int* __restrict__ gid, float* __restrict__ h,
                               float* __restrict__ pooled, int n4) {
  int i = blockIdx.x * blockDim.x + threadIdx.x;
  if (i >= n4) return;
  int row = (i * 4) / DIM;
  int c0  = (i * 4) & (DIM - 1);
  float4 v = *(const float4*)(x + (size_t)i * 4);
  float4 o;
  o.x = fmaxf(0.0f, v.x * ss[c0 + 0] + ss[DIM + c0 + 0]);
  o.y = fmaxf(0.0f, v.y * ss[c0 + 1] + ss[DIM + c0 + 1]);
  o.z = fmaxf(0.0f, v.z * ss[c0 + 2] + ss[DIM + c0 + 2]);
  o.w = fmaxf(0.0f, v.w * ss[c0 + 3] + ss[DIM + c0 + 3]);
  *(float4*)(h + (size_t)i * 4) = o;
  float* pb = pooled + (size_t)gid[row] * DIM + c0;
  atomicAdd(pb + 0, o.x); atomicAdd(pb + 1, o.y);
  atomicAdd(pb + 2, o.z); atomicAdd(pb + 3, o.w);
}

// plain pool (for hidden[0] = feat)
__global__ void k_pool(const float* __restrict__ x, const int* __restrict__ gid,
                       float* __restrict__ pooled, int n4) {
  int i = blockIdx.x * blockDim.x + threadIdx.x;
  if (i >= n4) return;
  int row = (i * 4) / DIM;
  int c0  = (i * 4) & (DIM - 1);
  float4 v = *(const float4*)(x + (size_t)i * 4);
  float* pb = pooled + (size_t)gid[row] * DIM + c0;
  atomicAdd(pb + 0, v.x); atomicAdd(pb + 1, v.y);
  atomicAdd(pb + 2, v.z); atomicAdd(pb + 3, v.w);
}

// ---------------- readout: score + log_softmax ----------------

__global__ void k_readout(const float* __restrict__ pooled, const float* __restrict__ Wp,
                          const float* __restrict__ bp, float* __restrict__ out) {
  int g = blockIdx.x * blockDim.x + threadIdx.x;
  if (g >= N_GRAPHS) return;
  float acc[N_OUT];
#pragma unroll
  for (int o = 0; o < N_OUT; ++o) acc[o] = 0.0f;
  for (int i = 0; i <= N_LAYERS; ++i) {
    const float* pg = pooled + ((size_t)i * N_GRAPHS + g) * DIM;
    const float* w  = Wp + (size_t)i * DIM * N_OUT;
    for (int f = 0; f < DIM; ++f) {
      float p = pg[f];
#pragma unroll
      for (int o = 0; o < N_OUT; ++o) acc[o] += p * w[f * N_OUT + o];
    }
#pragma unroll
    for (int o = 0; o < N_OUT; ++o) acc[o] += bp[i * N_OUT + o];
  }
  float m = acc[0];
#pragma unroll
  for (int o = 1; o < N_OUT; ++o) m = fmaxf(m, acc[o]);
  float se = 0.0f;
#pragma unroll
  for (int o = 0; o < N_OUT; ++o) se += expf(acc[o] - m);
  float lse = logf(se) + m;
#pragma unroll
  for (int o = 0; o < N_OUT; ++o) {
    out[(size_t)g * N_OUT + o] = acc[o] - lse;                        // log_softmax
    out[(size_t)N_GRAPHS * N_OUT + (size_t)g * N_OUT + o] = acc[o];   // raw score
  }
}

// ---------------- driver ----------------

extern "C" void kernel_launch(void* const* d_in, const int* in_sizes, int n_in,
                              void* d_out, int out_size, void* d_ws, size_t ws_size,
                              hipStream_t stream) {
  const float* feat  = (const float*)d_in[0];
  const float* W1    = (const float*)d_in[1];
  const float* W2    = (const float*)d_in[2];
  const float* g_mlp = (const float*)d_in[3];
  const float* b_mlp = (const float*)d_in[4];
  const float* g_out = (const float*)d_in[5];
  const float* b_out = (const float*)d_in[6];
  const float* Wp    = (const float*)d_in[7];
  const float* bp    = (const float*)d_in[8];
  const int*   src   = (const int*)d_in[9];
  const int*   dst   = (const int*)d_in[10];
  const int*   gid   = (const int*)d_in[11];

  char* ws = (char*)d_ws;
  const size_t NB = (size_t)N_NODES * DIM * sizeof(float);   // 51.2 MB
  float*          agg   = (float*)(ws);
  float*          z     = (float*)(ws + NB);
  float*          h     = (float*)(ws + 2 * NB);
  unsigned short* xb    = (unsigned short*)(ws + 3 * NB);    // 25.6 MB (xb & zb share)
  unsigned int*   wpk   = (unsigned int*)(ws + 3 * NB + NB / 2);
  float*          stats = (float*)(ws + 3 * NB + NB / 2 + 262144);
  float*          ss    = stats + 256;
  float*          pooled = ss + 256;                          // 5*512*128 f32

  const int n4 = N_NODES * DIM / 4;                           // 3,200,000
  const int EW_BLK = (n4 + 255) / 256;                        // 12500
  const int ROW_TILES = N_NODES / 16;                         // 6250
  const int GEMM_BLK = (ROW_TILES + 7) / 8;                   // 782

  // init pooled, swizzle weights, pool hidden[0]=feat
  k_zero<<<( (N_LAYERS + 1) * N_GRAPHS * DIM + 255) / 256, 256, 0, stream>>>(
      pooled, (N_LAYERS + 1) * N_GRAPHS * DIM);
  k_prep_w<<<256, 256, 0, stream>>>(W1, W2, wpk);
  k_pool<<<EW_BLK, 256, 0, stream>>>(feat, gid, pooled, n4);

  const float* hin = feat;
  for (int i = 0; i < N_LAYERS; ++i) {
    // agg = h + sum_{src->dst} h[src]
    k_copy4<<<EW_BLK, 256, 0, stream>>>((const float4*)hin, (float4*)agg, n4);
    k_edge_agg<<<N_EDGES / 8, 256, 0, stream>>>(hin, src, dst, agg);
    k_f32_to_bf16<<<EW_BLK, 256, 0, stream>>>(agg, xb, n4);

    // z = (h+agg) @ W1[i]  (WMMA bf16 -> f32)
    k_gemm_bf16<<<GEMM_BLK, 256, 0, stream>>>(xb, wpk + (size_t)(2 * i) * 8192, z, ROW_TILES);
    k_zero<<<1, 256, 0, stream>>>(stats, 256);
    k_bn_stats<<<512, 128, 0, stream>>>(z, stats);
    k_bn_finalize<<<1, 128, 0, stream>>>(stats, g_mlp + i * DIM, b_mlp + i * DIM, ss);
    k_bn_relu_bf16<<<EW_BLK, 256, 0, stream>>>(z, ss, xb, n4);   // zb reuses xb

    // h = relu(bn(z @ W2[i])), fused pooling into pooled[i+1]
    k_gemm_bf16<<<GEMM_BLK, 256, 0, stream>>>(xb, wpk + (size_t)(2 * i + 1) * 8192, z, ROW_TILES);
    k_zero<<<1, 256, 0, stream>>>(stats, 256);
    k_bn_stats<<<512, 128, 0, stream>>>(z, stats);
    k_bn_finalize<<<1, 128, 0, stream>>>(stats, g_out + i * DIM, b_out + i * DIM, ss);
    k_bn_relu_pool<<<EW_BLK, 256, 0, stream>>>(
        z, ss, gid, h, pooled + (size_t)(i + 1) * N_GRAPHS * DIM, n4);
    hin = h;
  }

  k_readout<<<(N_GRAPHS + 255) / 256, 256, 0, stream>>>(pooled, Wp, bp, (float*)d_out);
}
